// HiPPOLTI_28527172780058
// MI455X (gfx1250) — matrix operand order repaired and checked by reading, not compiled, not run on hardware
//
// HiPPO-LegS LTI scan on MI455X (gfx1250), restructured as a chunked parallel
// prefix so the 8192-step sequential recurrence becomes wide WMMA GEMM work.
//
//   c_t = Ad c_{t-1} + Bd f_t,   N=512, batch=8, L=8192, all states output.
//
// Passes:
//   1) local scan of J=32 steps for all 2048 (batch,chunk) chains -> u[chain]
//   2) Hillis-Steele prefix over 256 chunk boundaries with matrix operator
//      M^(2^d) = Ad^(32*2^d) (12 matrix squarings, 8 scan rounds, all GEMMs)
//   3) re-scan each chunk from its correct entry state, writing all outputs
//      (134 MB, the HBM floor ~5.8us at 23.3 TB/s) straight from accumulators.
//
// All matrix products use v_wmma_f32_16x16x32_bf16 with a bf16 hi/lo split
// (Ah*Bh + Ah*Bl + Al*Bh, f32 accumulate) for ~f32 accuracy at matrix-core
// rate. Ad (and all its powers) are lower-triangular -> strictly-upper K
// tiles are skipped (~2x less math and L2 traffic).
//
// All wave-uniform control (wave id, tile ids, triangular predicates) is
// forced into SGPRs via readfirstlane so every WMMA executes with EXEC
// all-ones (ISA 7.12 requirement) under scalar branches only.
//
// Workspace assumption: ws_size >= ~14 MB.

#include <hip/hip_runtime.h>
#include <stdint.h>

#define NN      512                    // state order
#define LSEQ    8192                   // sequence length
#define NBATCH  8
#define JC      32                     // chunk length (steps per chunk)
#define NCHUNK  (LSEQ / JC)            // 256 chunks per batch
#define CHAINS  (NBATCH * NCHUNK)      // 2048 independent local scans
#define ROUNDS  8                      // 2^8 == NCHUNK
#define PITCH   529                    // LDS pitch (odd, 17 mod 64 -> conflict-free)

typedef unsigned short u16;
typedef __attribute__((ext_vector_type(16))) __bf16   v16bf;
typedef __attribute__((ext_vector_type(8)))  float    v8f;
typedef __attribute__((ext_vector_type(4)))  unsigned u32x4;
typedef __attribute__((ext_vector_type(4)))  float    f32x4;

union ABOp {                 // one WMMA A or B operand (16x32 / 32x16 bf16)
  v16bf v;
  u32x4 q[2];
  u16   s[16];
};

// Split f32 into leading bf16 (truncated) + bf16 residual. hi is exact in the
// top 8 mantissa bits, lo captures the next ~8 -> product error ~2^-16.
static __device__ __forceinline__ void split_bf16(float x, u16& h, u16& l) {
  union { float f; unsigned u; } a, b, c;
  a.f = x;
  h   = (u16)(a.u >> 16);
  b.u = ((unsigned)h) << 16;
  c.f = x - b.f;
  l   = (u16)(c.u >> 16);
}

static __device__ __forceinline__ v8f vzero8() {
  v8f z = {0.f, 0.f, 0.f, 0.f, 0.f, 0.f, 0.f, 0.f};
  return z;
}

// 3-product bf16-split multiply-accumulate: (Ah+Al)(Bh+Bl) minus lo*lo term.
static __device__ __forceinline__ v8f wmma3(const ABOp& Ah, const ABOp& Al,
                                            const ABOp& Bh, const ABOp& Bl, v8f acc) {
  acc = __builtin_amdgcn_wmma_f32_16x16x32_bf16(false, Ah.v, false, Bh.v, (short)0, acc, false, false);
  acc = __builtin_amdgcn_wmma_f32_16x16x32_bf16(false, Ah.v, false, Bl.v, (short)0, acc, false, false);
  acc = __builtin_amdgcn_wmma_f32_16x16x32_bf16(false, Al.v, false, Bh.v, (short)0, acc, false, false);
  return acc;
}

// A-operand (16x32 bf16) per-lane load from row-major bf16 [NN][NN]:
// lane = 16h+m holds rows m, K-run [16h,16h+16) (contiguous, matches B)
// -> two global_load_b128 per operand.
static __device__ __forceinline__ void loadA(const u16* base, ABOp& op) {
  op.q[0] = *(const u32x4*)(base);
  op.q[1] = *(const u32x4*)(base + 8);
}

// Lower-triangular tile predicate: K-tile kt needed for M-tile mt iff
// kt*32 <= 16*mt+15.
static __device__ __forceinline__ int ktmax_of(int mt) { return (16 * mt + 15) >> 5; }

// Wave-uniform scalar (SGPR) copy of a value that is identical across a wave.
static __device__ __forceinline__ int uni(int x) {
  return __builtin_amdgcn_readfirstlane(x);
}

// ---------------------------------------------------------------------------
// Kernel 1: split Ad (f32) into row-major bf16 hi/lo planes.
// ---------------------------------------------------------------------------
__global__ __launch_bounds__(256) void split_kernel(const float* __restrict__ Ad,
                                                    u16* __restrict__ hi,
                                                    u16* __restrict__ lo) {
  for (int e = blockIdx.x * blockDim.x + threadIdx.x; e < NN * NN;
       e += gridDim.x * blockDim.x) {
    u16 h, l;
    split_bf16(Ad[e], h, l);
    hi[e] = h;
    lo[e] = l;
  }
}

// ---------------------------------------------------------------------------
// Kernel 2: matrix square C = A*A for lower-triangular A.
//   A given as bf16 hi/lo planes (WMMA A side) + an f32 copy (B side source).
//   Writes C as f32 plus fresh bf16 hi/lo planes for the next level.
//   Grid: 32 WGs, one 16-column block each; 8 waves x 4 M-tiles cover M=512.
// ---------------------------------------------------------------------------
__global__ __launch_bounds__(256) void matsq_kernel(const u16* __restrict__ Ahi,
                                                    const u16* __restrict__ Alo,
                                                    const float* __restrict__ Bf,
                                                    float* __restrict__ Cf,
                                                    u16* __restrict__ Chi,
                                                    u16* __restrict__ Clo) {
  const int tid  = threadIdx.x;
  const int lane = tid & 31;
  const int wv   = uni(tid >> 5);      // SGPR: wave id -> scalar tile control
  const int half = lane >> 4;
  const int ln   = lane & 15;
  const int n0   = blockIdx.x * 16;

  const int mts[4]  = {wv, 15 - wv, 16 + wv, 31 - wv};
  const int ktmx[4] = {ktmax_of(mts[0]), ktmax_of(mts[1]),
                       ktmax_of(mts[2]), ktmax_of(mts[3])};
  const int ktlim   = ktmx[3];         // largest M-tile this wave owns

  v8f acc[4] = {vzero8(), vzero8(), vzero8(), vzero8()};

  for (int kt = 0; kt <= ktlim; ++kt) {
    // Build B tile (32x16) from f32 columns of Bf: B[k][n], k=kt*32+16h+j, n=n0+ln.
    ABOp Bh, Bl;
    const float* bp = Bf + (size_t)(kt * 32 + half * 16) * NN + n0 + ln;
#pragma unroll
    for (int j = 0; j < 16; ++j) {
      u16 h, l;
      split_bf16(bp[(size_t)j * NN], h, l);
      Bh.s[j] = h;
      Bl.s[j] = l;
    }
#pragma unroll
    for (int t = 0; t < 4; ++t) {
      if (kt > ktmx[t]) continue;              // scalar branch (SGPR compare)
      const size_t aoff = (size_t)(16 * mts[t] + ln) * NN + kt * 32 + half * 16;
      ABOp Ah, Al;
      loadA(Ahi + aoff, Ah);
      loadA(Alo + aoff, Al);
      acc[t] = wmma3(Ah, Al, Bh, Bl, acc[t]);
    }
  }

  // D tile layout: VGPR v, lane 16h+ln -> M = 16*mt + 8h + v, N = n0 + ln.
#pragma unroll
  for (int t = 0; t < 4; ++t) {
    const int mbase = 16 * mts[t] + half * 8;
#pragma unroll
    for (int v = 0; v < 8; ++v) {
      const size_t idx = (size_t)(mbase + v) * NN + n0 + ln;
      const float  val = acc[t][v];
      Cf[idx] = val;
      u16 h, l;
      split_bf16(val, h, l);
      Chi[idx] = h;
      Clo[idx] = l;
    }
  }
}

// ---------------------------------------------------------------------------
// Kernel 3: one Hillis-Steele round over chunk boundaries:
//   Xout[chain] = Xin[chain] + P * Xin[chain - s]   (if chunk_index >= s)
// P = Ad^(JC * s) is lower-triangular, given as bf16 hi/lo planes.
// Grid: CHAINS/16 = 128 WGs; columns with chunk < s contribute zero operands
// (lane-divergent data only, never divergent control around WMMA).
// ---------------------------------------------------------------------------
__global__ __launch_bounds__(256) void round_kernel(const float* __restrict__ Xin,
                                                    float* __restrict__ Xout,
                                                    const u16* __restrict__ Phi,
                                                    const u16* __restrict__ Plo,
                                                    int shift) {
  const int tid  = threadIdx.x;
  const int lane = tid & 31;
  const int wv   = uni(tid >> 5);
  const int half = lane >> 4;
  const int ln   = lane & 15;

  const int chain = blockIdx.x * 16 + ln;
  const int c     = chain & (NCHUNK - 1);
  const bool has  = (c >= shift);
  // Clamp to a valid address so speculative loads stay in-bounds.
  const float* src = Xin + (size_t)(has ? (chain - shift) : chain) * NN;

  const int mts[4]  = {wv, 15 - wv, 16 + wv, 31 - wv};
  const int ktmx[4] = {ktmax_of(mts[0]), ktmax_of(mts[1]),
                       ktmax_of(mts[2]), ktmax_of(mts[3])};
  const int ktlim   = ktmx[3];

  v8f acc[4] = {vzero8(), vzero8(), vzero8(), vzero8()};

  for (int kt = 0; kt <= ktlim; ++kt) {
    ABOp Bh, Bl;
    const float* bp = src + kt * 32 + half * 16;
#pragma unroll
    for (int j = 0; j < 16; ++j) {
      const float x = has ? bp[j] : 0.f;       // lane-divergent data, not control
      u16 h, l;
      split_bf16(x, h, l);
      Bh.s[j] = h;
      Bl.s[j] = l;
    }
#pragma unroll
    for (int t = 0; t < 4; ++t) {
      if (kt > ktmx[t]) continue;              // scalar branch
      const size_t aoff = (size_t)(16 * mts[t] + ln) * NN + kt * 32 + half * 16;
      ABOp Ah, Al;
      loadA(Phi + aoff, Ah);
      loadA(Plo + aoff, Al);
      acc[t] = wmma3(Ah, Al, Bh, Bl, acc[t]);
    }
  }

#pragma unroll
  for (int t = 0; t < 4; ++t) {
    const int    mbase = 16 * mts[t] + half * 8;
    const size_t base  = (size_t)chain * NN + mbase;
    f32x4 a0 = *(const f32x4*)(Xin + base);
    f32x4 a1 = *(const f32x4*)(Xin + base + 4);
#pragma unroll
    for (int v = 0; v < 4; ++v) { a0[v] += acc[t][v]; a1[v] += acc[t][v + 4]; }
    *(f32x4*)(Xout + base)     = a0;
    *(f32x4*)(Xout + base + 4) = a1;
  }
}

// ---------------------------------------------------------------------------
// Kernel 4: chunk-local scan. Each WG owns 16 chains (columns) and the full
// 512-row state, kept transposed in LDS: S[n][k]. Per step:
//   S_new = Ad * S + Bd (x) f_step     (GEMM 512x512 x 512x16, triangular)
// MODE 0 (pass 1): init zero (chunk 0: init_state), emit only final state u.
// MODE 1 (pass 3): init from prefix Sprev, stream every state to `out` with
//                  b128 stores straight from the WMMA accumulators.
// ---------------------------------------------------------------------------
template <int MODE>
__global__ __launch_bounds__(256) void scan_kernel(const u16* __restrict__ Ahi,
                                                   const u16* __restrict__ Alo,
                                                   const float* __restrict__ Bd,
                                                   const float* __restrict__ f,
                                                   const float* __restrict__ init_state,
                                                   const float* __restrict__ Sprev,
                                                   float* __restrict__ Ufinal,
                                                   float* __restrict__ out) {
  __shared__ float S[16 * PITCH];   // S[n][k]: state k of column n (transposed)
  __shared__ float Fs[16 * JC];     // per-column inputs for this chunk
  __shared__ float BdS[NN];

  const int wg   = blockIdx.x;      // 0..127, 16 chains each
  const int tid  = threadIdx.x;
  const int lane = tid & 31;
  const int wv   = uni(tid >> 5);
  const int half = lane >> 4;
  const int ln   = lane & 15;

  for (int i = tid; i < NN; i += 256) BdS[i] = Bd[i];
  for (int i = tid; i < 16 * JC; i += 256) {
    const int n = i / JC, j = i % JC;
    const int chain = wg * 16 + n;
    const int b = chain >> 8, c = chain & (NCHUNK - 1);
    Fs[i] = f[(size_t)b * LSEQ + (size_t)c * JC + j];
  }
  for (int i = tid; i < 16 * NN; i += 256) {
    const int n = i >> 9, k = i & (NN - 1);
    const int chain = wg * 16 + n;
    const int b = chain >> 8, c = chain & (NCHUNK - 1);
    float v;
    if (c == 0)         v = init_state[(size_t)b * NN + k];
    else if (MODE == 0) v = 0.f;
    else                v = Sprev[(size_t)(chain - 1) * NN + k]; // s[c-1], same batch
    S[n * PITCH + k] = v;
  }
  __syncthreads();

  const int mts[4]  = {wv, 15 - wv, 16 + wv, 31 - wv};
  const int ktmx[4] = {ktmax_of(mts[0]), ktmax_of(mts[1]),
                       ktmax_of(mts[2]), ktmax_of(mts[3])};
  const int ktlim   = ktmx[3];

  // Warm L2 with this lane's Ad rows (global_prefetch_b8); Ad stays L2-resident.
#pragma unroll
  for (int t = 0; t < 4; ++t) {
    __builtin_prefetch(Ahi + (size_t)(16 * mts[t] + ln) * NN, 0, 1);
    __builtin_prefetch(Alo + (size_t)(16 * mts[t] + ln) * NN, 0, 1);
  }

  for (int j = 0; j < JC; ++j) {
    v8f acc[4] = {vzero8(), vzero8(), vzero8(), vzero8()};

    for (int kt = 0; kt <= ktlim; ++kt) {
      // B tile from transposed state: B[k][n] = S[n][k]; conflict-free pitch.
      ABOp Bh, Bl;
      const float* srow = &S[ln * PITCH + kt * 32 + half * 16];
#pragma unroll
      for (int jj = 0; jj < 16; ++jj) {
        u16 h, l;
        split_bf16(srow[jj], h, l);
        Bh.s[jj] = h;
        Bl.s[jj] = l;
      }
#pragma unroll
      for (int t = 0; t < 4; ++t) {
        if (kt > ktmx[t]) continue;            // scalar branch
        const size_t aoff = (size_t)(16 * mts[t] + ln) * NN + kt * 32 + half * 16;
        ABOp Ah, Al;
        loadA(Ahi + aoff, Ah);
        loadA(Alo + aoff, Al);
        acc[t] = wmma3(Ah, Al, Bh, Bl, acc[t]);
      }
    }
    __syncthreads();  // all reads of S complete before overwrite

    const float fv    = Fs[ln * JC + j];     // this column's input at step j
    const int   chain = wg * 16 + ln;
    const int   b     = chain >> 8;
    const int   c     = chain & (NCHUNK - 1);
    const size_t tabs = (size_t)c * JC + j;

#pragma unroll
    for (int t = 0; t < 4; ++t) {
      const int mbase = 16 * mts[t] + half * 8;
      float vals[8];
#pragma unroll
      for (int v = 0; v < 8; ++v) vals[v] = acc[t][v] + BdS[mbase + v] * fv;
#pragma unroll
      for (int v = 0; v < 8; ++v) S[ln * PITCH + mbase + v] = vals[v];
      if (MODE == 1) {
        float* op = out + ((size_t)b * LSEQ + tabs) * NN + mbase;
        f32x4 s0 = {vals[0], vals[1], vals[2], vals[3]};
        f32x4 s1 = {vals[4], vals[5], vals[6], vals[7]};
        *(f32x4*)(op)     = s0;
        *(f32x4*)(op + 4) = s1;
      }
    }
    __syncthreads();  // S fully updated before next step's reads
  }

  if (MODE == 0) {
    for (int i = tid; i < 16 * NN; i += 256) {
      const int n = i >> 9, k = i & (NN - 1);
      Ufinal[(size_t)(wg * 16 + n) * NN + k] = S[n * PITCH + k];
    }
  }
}

// ---------------------------------------------------------------------------
// Host orchestration: 1 split + 12 squarings + pass1 + 8 rounds + pass3,
// all on `stream`, graph-capture safe (no allocs/syncs).
// ---------------------------------------------------------------------------
extern "C" void kernel_launch(void* const* d_in, const int* in_sizes, int n_in,
                              void* d_out, int out_size, void* d_ws, size_t ws_size,
                              hipStream_t stream) {
  const float* f    = (const float*)d_in[0];   // (8, 8192, 1)
  const float* init = (const float*)d_in[1];   // (8, 1, 512)
  const float* Ad   = (const float*)d_in[2];   // (512, 512)
  const float* Bd   = (const float*)d_in[3];   // (512, 1)
  float*       out  = (float*)d_out;           // (8, 8192, 1, 512)

  uint8_t* w = (uint8_t*)d_ws;
  auto take = [&](size_t bytes) -> void* {
    void* p = (void*)w;
    w += (bytes + 255) & ~(size_t)255;
    return p;
  };
  u16*   Ahi = (u16*)take((size_t)NN * NN * 2);
  u16*   Alo = (u16*)take((size_t)NN * NN * 2);
  float* Uf  = (float*)take((size_t)NN * NN * 4);
  u16*   Uhi = (u16*)take((size_t)NN * NN * 2);
  u16*   Ulo = (u16*)take((size_t)NN * NN * 2);
  float* Vf  = (float*)take((size_t)NN * NN * 4);
  u16*   Vhi = (u16*)take((size_t)NN * NN * 2);
  u16*   Vlo = (u16*)take((size_t)NN * NN * 2);
  float* X0  = (float*)take((size_t)CHAINS * NN * 4);
  float* X1  = (float*)take((size_t)CHAINS * NN * 4);
  (void)in_sizes; (void)n_in; (void)out_size; (void)ws_size;  // ~13.3 MB used

  split_kernel<<<256, 256, 0, stream>>>(Ad, Ahi, Alo);

  // Squaring ladder up to M = Ad^JC = Ad^32 (ends in U).
  matsq_kernel<<<32, 256, 0, stream>>>(Ahi, Alo, Ad, Uf, Uhi, Ulo);  // Ad^2
  matsq_kernel<<<32, 256, 0, stream>>>(Uhi, Ulo, Uf, Vf, Vhi, Vlo);  // Ad^4
  matsq_kernel<<<32, 256, 0, stream>>>(Vhi, Vlo, Vf, Uf, Uhi, Ulo);  // Ad^8
  matsq_kernel<<<32, 256, 0, stream>>>(Uhi, Ulo, Uf, Vf, Vhi, Vlo);  // Ad^16
  matsq_kernel<<<32, 256, 0, stream>>>(Vhi, Vlo, Vf, Uf, Uhi, Ulo);  // Ad^32 = P0

  // Pass 1: local chunk scans -> u in X0.
  scan_kernel<0><<<128, 256, 0, stream>>>(Ahi, Alo, Bd, f, init, X0, X0, out);

  // Hillis-Steele prefix over chunk boundaries, squaring P between rounds.
  float* curF = Uf;  u16* curH = Uhi;  u16* curL = Ulo;
  float* othF = Vf;  u16* othH = Vhi;  u16* othL = Vlo;
  float* xin = X0;   float* xout = X1;
  for (int d = 0; d < ROUNDS; ++d) {
    round_kernel<<<128, 256, 0, stream>>>(xin, xout, curH, curL, 1 << d);
    if (d < ROUNDS - 1) {
      matsq_kernel<<<32, 256, 0, stream>>>(curH, curL, curF, othF, othH, othL);
      float* tf = curF; curF = othF; othF = tf;
      u16*   th = curH; curH = othH; othH = th;
      u16*   tl = curL; curL = othL; othL = tl;
    }
    float* tx = xin; xin = xout; xout = tx;
  }
  // Inclusive prefix s[c] now lives where `xin` points (ROUNDS even -> X0).

  // Pass 3: full re-scan from correct entry states, writing all outputs.
  scan_kernel<1><<<128, 256, 0, stream>>>(Ahi, Alo, Bd, f, init, xin, X1, out);
}
